// AttForward_47631187313264
// MI455X (gfx1250) — matrix operand order repaired and checked by reading, not compiled, not run on hardware
//
#include <hip/hip_runtime.h>
#include <hip/hip_bf16.h>
#include <math.h>

// Shapes from the reference
#define BB 64
#define TT 1024
#define EPROJS 512
#define DUNITS 1024
#define ATTD 512
#define CHANS 32
#define FILTS 15
#define KW 31

typedef __attribute__((ext_vector_type(16))) __bf16 v16bf;
typedef __attribute__((ext_vector_type(8)))  float  v8f;

// ---------------- workspace layout (bytes) ----------------
#define O_WENC   ((size_t)0)                         // 512 frags * 32 lanes * 16 bf16 = 512KB
#define O_WATT   ((size_t)(O_WENC + 524288))         // 32 frags  * 32 * 16 bf16      = 32KB
#define O_DB     ((size_t)(O_WATT + 32768))          // 64*512 f32                    = 128KB
#define O_CONV   ((size_t)(O_DB   + 131072))         // 65536*32 bf16                 = 4MB
#define O_E      ((size_t)(O_CONV + 4194304))        // 65536 f32                     = 256KB
#define O_W      ((size_t)(O_E    + 262144))         // 65536 f32                     = 256KB

// ---------------------------------------------------------------------------
// Pack W_enc (512x512 row-major, K-major) into per-lane WMMA B fragments.
// Fragment index f = n_tile*16 + k_tile. Lane layout: col n = n_tile*16 + (lane&15),
// K = k_tile*32 + (lane>>4)*16 + j for j=0..15.
// ---------------------------------------------------------------------------
__global__ __launch_bounds__(256) void k_pack_wenc(const float* __restrict__ W,
                                                   __bf16* __restrict__ dst) {
  int lane = threadIdx.x & 31;
  int wave = blockIdx.x * 8 + (threadIdx.x >> 5);   // 64 blocks * 8 waves = 512 frags
  int n_tile = wave >> 4;
  int k_tile = wave & 15;
  int n = n_tile * 16 + (lane & 15);
  int kbase = k_tile * 32 + (lane >> 4) * 16;
  __bf16* p = dst + ((size_t)wave * 32 + lane) * 16;
#pragma unroll
  for (int j = 0; j < 16; ++j)
    p[j] = (__bf16)W[(size_t)(kbase + j) * ATTD + n];
}

// W_att (32 x 512): one fragment per n_tile (K = 32 channels)
__global__ __launch_bounds__(256) void k_pack_watt(const float* __restrict__ W,
                                                   __bf16* __restrict__ dst) {
  int lane = threadIdx.x & 31;
  int wave = blockIdx.x * 8 + (threadIdx.x >> 5);   // 4 blocks * 8 = 32 frags
  int n = wave * 16 + (lane & 15);
  int kbase = (lane >> 4) * 16;
  __bf16* p = dst + ((size_t)wave * 32 + lane) * 16;
#pragma unroll
  for (int j = 0; j < 16; ++j)
    p[j] = (__bf16)W[(size_t)(kbase + j) * ATTD + n];
}

// ---------------------------------------------------------------------------
// db[b][a] = b_enc[a] + dec_z[b,:] @ W_dec[:,a]   (64 x 512)
// ---------------------------------------------------------------------------
__global__ __launch_bounds__(256) void k_dec(const float* __restrict__ dec_z,
                                             const float* __restrict__ W_dec,
                                             const float* __restrict__ b_enc,
                                             float* __restrict__ db) {
  int idx = blockIdx.x * 256 + threadIdx.x;         // 128 blocks -> 32768
  int b = idx >> 9;
  int a = idx & 511;
  float s = b_enc[a];
  const float* dz = dec_z + (size_t)b * DUNITS;
  for (int k = 0; k < DUNITS; ++k)
    s = fmaf(dz[k], W_dec[(size_t)k * ATTD + a], s);
  db[idx] = s;
}

// ---------------------------------------------------------------------------
// Location conv: conv[bt][c] = sum_j att_prev[b, t-15+j] * conv_w[c][j]
// Output in (bt, 32) bf16 (A-fragment friendly: contiguous channels per row).
// Block covers 8 t-values x 32 channels; att chunk + weights staged in LDS.
// ---------------------------------------------------------------------------
__global__ __launch_bounds__(256) void k_conv(const float* __restrict__ att_prev,
                                              const float* __restrict__ conv_w,
                                              __bf16* __restrict__ out) {
  __shared__ float ap_s[8 + KW - 1];                // 38
  __shared__ float cw_s[CHANS * KW];                // 992
  int b  = blockIdx.x >> 7;                         // 8192 blocks = 64 * 128
  int t0 = (blockIdx.x & 127) * 8;
  for (int i = threadIdx.x; i < CHANS * KW; i += 256) cw_s[i] = conv_w[i];
  for (int i = threadIdx.x; i < 8 + KW - 1; i += 256) {
    int t = t0 - FILTS + i;
    ap_s[i] = (t >= 0 && t < TT) ? att_prev[(size_t)b * TT + t] : 0.f;
  }
  __syncthreads();
  int tl = threadIdx.x >> 5;                        // 0..7
  int c  = threadIdx.x & 31;
  float s = 0.f;
#pragma unroll
  for (int j = 0; j < KW; ++j)
    s = fmaf(ap_s[tl + j], cw_s[c * KW + j], s);
  out[((size_t)(b * TT + t0 + tl)) * CHANS + c] = (__bf16)s;
}

// ---------------------------------------------------------------------------
// Fused main kernel: per 16-row tile (rows = flattened b*T+t):
//   x[m,n] = enc[m,:] @ W_enc[:,n]  (16 K-steps of wmma bf16)
//          + conv[m,:] @ W_att[:,n] (1 extra wmma, K=32 channels)
//          + db[b][n]
//   e[m]  = sum_n tanh(x[m,n]) * w_g[n] + b_g
// enc read once from HBM; W_enc packed (512KB) streams from L2.
// ---------------------------------------------------------------------------
__global__ __launch_bounds__(256) void k_fused_e(const float* __restrict__ enc,
                                                 const __bf16* __restrict__ convb,
                                                 const __bf16* __restrict__ wencp,
                                                 const __bf16* __restrict__ wattp,
                                                 const float* __restrict__ db,
                                                 const float* __restrict__ w_g,
                                                 const float* __restrict__ b_g,
                                                 float* __restrict__ e_out) {
  int lane = threadIdx.x & 31;
  int wave = blockIdx.x * (blockDim.x >> 5) + (threadIdx.x >> 5);
  int m0   = wave << 4;                 // row-tile base (4096 tiles total)
  int b    = m0 >> 10;                  // batch index (tile never crosses b)
  int rlow = lane & 15;
  int hi   = lane >> 4;
  int klo  = hi << 4;                   // A-fragment K offset (0 or 16)
  int row  = m0 + rlow;

  // ---- load 16 enc A-fragments (16x32 bf16 each) into VGPRs ----
  const float* arow = enc + (size_t)row * EPROJS + klo;
  v16bf afr[16];
#pragma unroll
  for (int kt = 0; kt < 16; ++kt) {
    const float4* p = (const float4*)(arow + kt * 32);
    float4 f0 = p[0], f1 = p[1], f2 = p[2], f3 = p[3];
    float tmp[16] = {f0.x, f0.y, f0.z, f0.w, f1.x, f1.y, f1.z, f1.w,
                     f2.x, f2.y, f2.z, f2.w, f3.x, f3.y, f3.z, f3.w};
#pragma unroll
    for (int j = 0; j < 16; ++j) afr[kt][j] = (__bf16)tmp[j];
  }
  // ---- conv A-fragment (16 rows x 32 channels) ----
  const __bf16* crow = convb + (size_t)row * CHANS + klo;
  v16bf acv;
#pragma unroll
  for (int j = 0; j < 8; ++j) { acv[j] = crow[j]; acv[8 + j] = crow[8 + j]; }

  float esum[8] = {0.f, 0.f, 0.f, 0.f, 0.f, 0.f, 0.f, 0.f};

  for (int nt = 0; nt < 32; ++nt) {
    v8f acc = {0.f, 0.f, 0.f, 0.f, 0.f, 0.f, 0.f, 0.f};
    const __bf16* bp = wencp + ((size_t)(nt * 16) * 32 + lane) * 16;
    __builtin_prefetch(bp + (size_t)16 * 32 * 16, 0, 3);   // next n-tile's B panel
#pragma unroll
    for (int kt = 0; kt < 16; ++kt) {
      v16bf bf = *(const v16bf*)(bp + (size_t)kt * 32 * 16);
      acc = __builtin_amdgcn_wmma_f32_16x16x32_bf16(false, afr[kt], false, bf,
                                                    (short)0, acc, false, false);
    }
    v16bf ba = *(const v16bf*)(wattp + ((size_t)nt * 32 + lane) * 16);
    acc = __builtin_amdgcn_wmma_f32_16x16x32_bf16(false, acv, false, ba,
                                                  (short)0, acc, false, false);
    int n = (nt << 4) + rlow;
    float dbv = db[(size_t)b * ATTD + n];
    float wgv = w_g[n];
#pragma unroll
    for (int r = 0; r < 8; ++r) {
      float x = acc[r] + dbv;             // C layout: lanes0-15 -> M=r, lanes16-31 -> M=8+r
      esum[r] += tanhf(x) * wgv;
    }
  }

  float bg = b_g[0];
#pragma unroll
  for (int r = 0; r < 8; ++r) {
    float v = esum[r];
#pragma unroll
    for (int o = 8; o > 0; o >>= 1) v += __shfl_xor(v, o, 32);  // within 16-lane half
    if (rlow == 0) e_out[m0 + hi * 8 + r] = v + bg;
  }
}

// ---------------------------------------------------------------------------
// Per batch row: mask, softmax, forward recurrence, clip, L1 normalize.
// ---------------------------------------------------------------------------
__global__ __launch_bounds__(1024) void k_softmax(const float* __restrict__ e_in,
                                                  const float* __restrict__ att_prev,
                                                  const int* __restrict__ len,
                                                  float* __restrict__ out_w,
                                                  float* __restrict__ ws_w) {
  __shared__ float red[32];
  int b = blockIdx.x, t = threadIdx.x;
  int lane = t & 31, wid = t >> 5;
  int L = len[b];
  float ev = (t < L) ? e_in[(size_t)b * TT + t] : -INFINITY;

  float v = ev;
#pragma unroll
  for (int o = 16; o > 0; o >>= 1) v = fmaxf(v, __shfl_xor(v, o, 32));
  if (lane == 0) red[wid] = v;
  __syncthreads();
  float m = red[0];
  for (int i = 1; i < 32; ++i) m = fmaxf(m, red[i]);
  __syncthreads();

  float ex = __expf(ev - m);               // exp(-inf) -> 0 for masked
  v = ex;
#pragma unroll
  for (int o = 16; o > 0; o >>= 1) v += __shfl_xor(v, o, 32);
  if (lane == 0) red[wid] = v;
  __syncthreads();
  float s = 0.f;
  for (int i = 0; i < 32; ++i) s += red[i];
  __syncthreads();
  float wsoft = ex / s;

  float ap  = att_prev[(size_t)b * TT + t];
  float aps = t ? att_prev[(size_t)b * TT + t - 1] : 0.f;
  float ww  = fmaxf((ap + aps) * wsoft, 1e-6f);

  v = ww;
#pragma unroll
  for (int o = 16; o > 0; o >>= 1) v += __shfl_xor(v, o, 32);
  if (lane == 0) red[wid] = v;
  __syncthreads();
  float tot = 0.f;
  for (int i = 0; i < 32; ++i) tot += red[i];

  float wn = ww / tot;
  out_w[(size_t)b * TT + t] = wn;
  ws_w[(size_t)b * TT + t]  = wn;
}

// ---------------------------------------------------------------------------
// Context: c[b,e] = sum_t enc[b,t,e] * w[b,t]   (second & final enc read)
// ---------------------------------------------------------------------------
__global__ __launch_bounds__(512) void k_context(const float* __restrict__ enc,
                                                 const float* __restrict__ w,
                                                 float* __restrict__ c_out) {
  __shared__ float ws_[TT];
  int b = blockIdx.x, e = threadIdx.x;
  for (int i = threadIdx.x; i < TT; i += 512) ws_[i] = w[(size_t)b * TT + i];
  __syncthreads();
  const float* p = enc + ((size_t)b * TT) * EPROJS + e;
  float acc = 0.f;
  for (int t = 0; t < TT; ++t) acc = fmaf(p[(size_t)t * EPROJS], ws_[t], acc);
  c_out[(size_t)b * EPROJS + e] = acc;
}

// ---------------------------------------------------------------------------
extern "C" void kernel_launch(void* const* d_in, const int* in_sizes, int n_in,
                              void* d_out, int out_size, void* d_ws, size_t ws_size,
                              hipStream_t stream) {
  const float* enc_hs   = (const float*)d_in[0];
  const int*   enc_len  = (const int*)  d_in[1];
  const float* dec_z    = (const float*)d_in[2];
  const float* att_prev = (const float*)d_in[3];
  const float* W_enc    = (const float*)d_in[4];
  const float* b_enc    = (const float*)d_in[5];
  const float* W_dec    = (const float*)d_in[6];
  const float* W_att    = (const float*)d_in[7];
  const float* conv_w   = (const float*)d_in[8];
  const float* w_g      = (const float*)d_in[9];
  const float* b_g      = (const float*)d_in[10];

  char* ws = (char*)d_ws;
  __bf16* wencp = (__bf16*)(ws + O_WENC);
  __bf16* wattp = (__bf16*)(ws + O_WATT);
  float*  db    = (float*) (ws + O_DB);
  __bf16* convb = (__bf16*)(ws + O_CONV);
  float*  e_buf = (float*) (ws + O_E);
  float*  w_buf = (float*) (ws + O_W);

  float* c_out = (float*)d_out;                    // (64, 512)
  float* w_out = (float*)d_out + BB * EPROJS;      // (64, 1024)

  k_pack_wenc<<<64, 256, 0, stream>>>(W_enc, wencp);
  k_pack_watt<<<4, 256, 0, stream>>>(W_att, wattp);
  k_dec<<<128, 256, 0, stream>>>(dec_z, W_dec, b_enc, db);
  k_conv<<<BB * (TT / 8), 256, 0, stream>>>(att_prev, conv_w, convb);
  k_fused_e<<<512, 256, 0, stream>>>(enc_hs, convb, wencp, wattp, db, w_g, b_g, e_buf);
  k_softmax<<<BB, 1024, 0, stream>>>(e_buf, att_prev, enc_len, w_out, w_buf);
  k_context<<<BB, 512, 0, stream>>>(enc_hs, w_buf, c_out);
}
